// Block_5875515261621
// MI455X (gfx1250) — matrix operand order repaired and checked by reading, not compiled, not run on hardware
//
#include <hip/hip_runtime.h>
#include <hip/hip_bf16.h>

typedef __attribute__((ext_vector_type(16))) _Float16 v16h;
typedef __attribute__((ext_vector_type(8)))  _Float16 v8h;
typedef __attribute__((ext_vector_type(8)))  float    v8f;
typedef __attribute__((ext_vector_type(4)))  int      v4i;

#define NB   2
#define TT   4096
#define CC   512
#define NH   8
#define HDIM 64
#define MR   (NB*TT)   // 8192 rows

#if __has_builtin(__builtin_amdgcn_global_load_async_to_lds_b128)
#define HAVE_ASYNC 1
typedef __attribute__((address_space(1))) v4i gv4i;
typedef __attribute__((address_space(3))) v4i lv4i;
static __device__ __forceinline__ void async_cp16(const _Float16* g, _Float16* l) {
  __builtin_amdgcn_global_load_async_to_lds_b128((gv4i*)g, (lv4i*)l, 0, 0);
}
static __device__ __forceinline__ void async_wait0() {
#if __has_builtin(__builtin_amdgcn_s_wait_asynccnt)
  __builtin_amdgcn_s_wait_asynccnt(0);
#else
  asm volatile("s_wait_asynccnt 0" ::: "memory");
#endif
}
#else
#define HAVE_ASYNC 0
#endif

static __device__ __forceinline__ v16h mk16(v8h a, v8h b) {
  return __builtin_shufflevector(a, b, 0,1,2,3,4,5,6,7,8,9,10,11,12,13,14,15);
}

// 0.5*x*(1+tanh(u)) == x * e^{2u} / (e^{2u}+1), branch-free (v_exp + v_rcp)
static __device__ __forceinline__ float gelu_tanh(float x) {
  float u = 0.7978845608028654f * (x + 0.044715f * x * x * x);
  float t = __expf(fminf(2.0f * u, 80.0f));
  return x * t * __builtin_amdgcn_rcpf(t + 1.0f);
}

// ---------------------------------------------------------------------------
// f32 -> f16 convert (weights)
// ---------------------------------------------------------------------------
__global__ void cvt_kernel(const float* __restrict__ in, _Float16* __restrict__ out, int n) {
  int i = blockIdx.x * blockDim.x + threadIdx.x;
  int stride = gridDim.x * blockDim.x;
  for (; i < n; i += stride) out[i] = (_Float16)in[i];
}

// ---------------------------------------------------------------------------
// LayerNorm over C=512, f32 in -> f16 out.  256 threads, 2 elems/thread.
// ---------------------------------------------------------------------------
__global__ __launch_bounds__(256) void ln_kernel(const float* __restrict__ x,
                                                 const float* __restrict__ g,
                                                 const float* __restrict__ b,
                                                 _Float16* __restrict__ out) {
  int row = blockIdx.x;
  int tid = threadIdx.x;
  __shared__ float red[256];
  const float* xr = x + (size_t)row * CC;
  float v0 = xr[tid], v1 = xr[tid + 256];
  red[tid] = v0 + v1;
  __syncthreads();
  #pragma unroll
  for (int off = 128; off > 0; off >>= 1) {
    if (tid < off) red[tid] += red[tid + off];
    __syncthreads();
  }
  float mu = red[0] * (1.0f / CC);
  __syncthreads();
  float d0 = v0 - mu, d1 = v1 - mu;
  red[tid] = d0 * d0 + d1 * d1;
  __syncthreads();
  #pragma unroll
  for (int off = 128; off > 0; off >>= 1) {
    if (tid < off) red[tid] += red[tid + off];
    __syncthreads();
  }
  float rstd = rsqrtf(red[0] * (1.0f / CC) + 1e-5f);
  _Float16* orow = out + (size_t)row * CC;
  orow[tid]       = (_Float16)(d0 * rstd * g[tid]       + b[tid]);
  orow[tid + 256] = (_Float16)(d1 * rstd * g[tid + 256] + b[tid + 256]);
}

// ---------------------------------------------------------------------------
// WMMA GEMM:  out = act(A[MxK] * B[KxN] + bias) (+resid)
// Workgroup tile 128x128, 8 waves; wave = 32 rows x 64 cols (2x4 subtiles,
// 8 wmma / K-step).  K-step 32.  DOUBLE-BUFFERED LDS: tile t+1 is staged
// (async global->LDS for A) while tile t runs on the WMMA pipe; one
// s_wait_asynccnt + one barrier per K-step.
// ---------------------------------------------------------------------------
__global__ __launch_bounds__(256) void gemm_kernel(
    const _Float16* __restrict__ A, const _Float16* __restrict__ Bw,
    const float* __restrict__ bias, const float* __restrict__ resid,
    float* __restrict__ outF, _Float16* __restrict__ outH,
    int M, int N, int K, int act_gelu, float qscale, int qcols) {
  __shared__ __align__(16) _Float16 As[2][128 * 32];   // [m][k]
  __shared__ __align__(16) _Float16 Bs[2][128 * 32];   // [n][k] (transposed)

  int tid   = threadIdx.x;
  int lane  = tid & 31;
  int wave  = tid >> 5;
  int ln    = lane & 15;
  int h2    = lane >> 4;
  int mwave = wave & 3;    // 4 wave rows of 32
  int nwave = wave >> 2;   // 2 wave cols of 64
  int m0 = blockIdx.x * 128;
  int n1 = blockIdx.y * 128;

  auto stage_tile = [&](int k0, int buf) {
    // A tile (128x32): 16B chunks, async to LDS when available
    #pragma unroll
    for (int i = 0; i < 2; ++i) {
      int c = tid + i * 256;            // chunk id [0,512)
      int m = c >> 2, kk = (c & 3) * 8;
      const _Float16* gsrc = A + (size_t)(m0 + m) * K + (k0 + kk);
#if HAVE_ASYNC
      async_cp16(gsrc, &As[buf][c * 8]);
#else
      *(v8h*)&As[buf][c * 8] = *(const v8h*)gsrc;
#endif
    }
    // B tile transposed: Bs[n][k] = Bw[k0+k][n1+n]
    #pragma unroll
    for (int i = 0; i < 2; ++i) {
      int c = tid + i * 256;            // chunk id [0,512)
      int kk = c >> 4, n8 = (c & 15) * 8;
      v8h bv = *(const v8h*)(Bw + (size_t)(k0 + kk) * N + (n1 + n8));
      #pragma unroll
      for (int e = 0; e < 8; ++e) Bs[buf][(n8 + e) * 32 + kk] = bv[e];
    }
  };

  v8f acc[2][4];
  v8f zero = {};
  #pragma unroll
  for (int i = 0; i < 2; ++i)
    #pragma unroll
    for (int j = 0; j < 4; ++j) acc[i][j] = zero;

  const int nT = K >> 5;

  // prologue: stage tile 0
  stage_tile(0, 0);
#if HAVE_ASYNC
  async_wait0();
#endif
  __syncthreads();

  for (int t = 0; t < nT; ++t) {
    int buf = t & 1;

    // 1) issue this tile's fragment loads (in-order DS queue)
    v16h af[2], bf[4];
    #pragma unroll
    for (int i = 0; i < 2; ++i) {
      const _Float16* ab = As[buf] + (mwave * 32 + i * 16 + ln) * 32;
      af[i] = mk16(*(const v8h*)(ab + 16 * h2), *(const v8h*)(ab + 16 * h2 + 8));
    }
    #pragma unroll
    for (int j = 0; j < 4; ++j) {
      const _Float16* bb = Bs[buf] + (nwave * 64 + j * 16 + ln) * 32 + 16 * h2;
      bf[j] = mk16(*(const v8h*)bb, *(const v8h*)(bb + 8));
    }

    // 2) kick off staging of tile t+1 into the other buffer
    if (t + 1 < nT) stage_tile((t + 1) << 5, buf ^ 1);

    // 3) WMMA on tile t
    #pragma unroll
    for (int i = 0; i < 2; ++i)
      #pragma unroll
      for (int j = 0; j < 4; ++j)
        acc[i][j] = __builtin_amdgcn_wmma_f32_16x16x32_f16(false, af[i], false, bf[j],
                                                           (short)0, acc[i][j], false, false);

    // 4) close the pipeline stage
    if (t + 1 < nT) {
#if HAVE_ASYNC
      async_wait0();
#endif
      __syncthreads();
    }
  }

  // ---- epilogue ----
  #pragma unroll
  for (int i = 0; i < 2; ++i) {
    #pragma unroll
    for (int j = 0; j < 4; ++j) {
      #pragma unroll
      for (int r = 0; r < 8; ++r) {
        int row = m0 + mwave * 32 + i * 16 + r + 8 * h2;
        int col = n1 + nwave * 64 + j * 16 + ln;
        float val = acc[i][j][r] + bias[col];
        if (act_gelu) val = gelu_tanh(val);
        if (resid) val += resid[(size_t)row * N + col];
        if (outF) outF[(size_t)row * N + col] = val;
        if (outH) {
          float w = (col < qcols) ? val * qscale : val;
          outH[(size_t)row * N + col] = (_Float16)w;
        }
      }
    }
  }
}

// ---------------------------------------------------------------------------
// Flash attention, causal.  qkv f16 [B*T][3C], q|k|v at col 0|512|1024,
// head h occupies 64 cols.  One wave per (16-row tile, head, batch).
// q pre-scaled by 1/sqrt(C) at qkv store time.
// ---------------------------------------------------------------------------
__global__ __launch_bounds__(32) void attn_kernel(const _Float16* __restrict__ qkv,
                                                  _Float16* __restrict__ y) {
  __shared__ __align__(16) _Float16 Pt[16 * 32];   // P tile [m][s]
  __shared__ __align__(16) _Float16 Vt[64 * 32];   // V transposed [c][s]

  int m0 = blockIdx.x * 16;
  int h  = blockIdx.y;
  int b  = blockIdx.z;
  size_t bT = (size_t)b * TT;
  int lane = threadIdx.x;
  int ln = lane & 15;
  int h2 = lane >> 4;

  const _Float16* qp = qkv + (bT + m0 + ln) * (3 * CC) + h * HDIM;
  v16h qf0 = mk16(*(const v8h*)(qp + 16 * h2),      *(const v8h*)(qp + 16 * h2 + 8));
  v16h qf1 = mk16(*(const v8h*)(qp + 32 + 16 * h2), *(const v8h*)(qp + 32 + 16 * h2 + 8));

  float rmax[8], rsum[8];
  v8f o[4];
  v8f zero = {};
  #pragma unroll
  for (int r = 0; r < 8; ++r) { rmax[r] = -1e30f; rsum[r] = 0.0f; }
  #pragma unroll
  for (int nt = 0; nt < 4; ++nt) o[nt] = zero;

  int sEnd = m0 + 16;
  for (int s0 = 0; s0 < sEnd; s0 += 32) {
    v8f sa0 = zero, sa1 = zero;
    {
      const _Float16* kp = qkv + (bT + s0 + ln) * (3 * CC) + CC + h * HDIM;
      v16h k0 = mk16(*(const v8h*)(kp + 16 * h2),      *(const v8h*)(kp + 16 * h2 + 8));
      v16h k1 = mk16(*(const v8h*)(kp + 32 + 16 * h2), *(const v8h*)(kp + 32 + 16 * h2 + 8));
      sa0 = __builtin_amdgcn_wmma_f32_16x16x32_f16(false, qf0, false, k0, (short)0, sa0, false, false);
      sa0 = __builtin_amdgcn_wmma_f32_16x16x32_f16(false, qf1, false, k1, (short)0, sa0, false, false);
    }
    {
      const _Float16* kp = qkv + (bT + s0 + 16 + ln) * (3 * CC) + CC + h * HDIM;
      v16h k0 = mk16(*(const v8h*)(kp + 16 * h2),      *(const v8h*)(kp + 16 * h2 + 8));
      v16h k1 = mk16(*(const v8h*)(kp + 32 + 16 * h2), *(const v8h*)(kp + 32 + 16 * h2 + 8));
      sa1 = __builtin_amdgcn_wmma_f32_16x16x32_f16(false, qf0, false, k0, (short)0, sa1, false, false);
      sa1 = __builtin_amdgcn_wmma_f32_16x16x32_f16(false, qf1, false, k1, (short)0, sa1, false, false);
    }

    if (s0 + 31 > m0) {                 // uniform branch; arithmetic mask
      #pragma unroll
      for (int r = 0; r < 8; ++r) {
        int row = m0 + r + 8 * h2;
        if (s0 + ln > row)      sa0[r] = -1e30f;
        if (s0 + 16 + ln > row) sa1[r] = -1e30f;
      }
    }

    float p0[8], p1[8];
    #pragma unroll
    for (int r = 0; r < 8; ++r) {
      float tmax = fmaxf(sa0[r], sa1[r]);
      #pragma unroll
      for (int msk = 1; msk < 16; msk <<= 1)
        tmax = fmaxf(tmax, __shfl_xor(tmax, msk, 16));
      float nm = fmaxf(rmax[r], tmax);
      float corr = __expf(rmax[r] - nm);
      p0[r] = __expf(sa0[r] - nm);
      p1[r] = __expf(sa1[r] - nm);
      float ps = p0[r] + p1[r];
      #pragma unroll
      for (int msk = 1; msk < 16; msk <<= 1)
        ps += __shfl_xor(ps, msk, 16);
      rsum[r] = rsum[r] * corr + ps;
      rmax[r] = nm;
      #pragma unroll
      for (int nt = 0; nt < 4; ++nt) o[nt][r] *= corr;
    }

    __syncthreads();
    #pragma unroll
    for (int r = 0; r < 8; ++r) {
      Pt[(r + 8 * h2) * 32 + ln]      = (_Float16)p0[r];
      Pt[(r + 8 * h2) * 32 + 16 + ln] = (_Float16)p1[r];
    }
    {
      const _Float16* vp = qkv + (bT + s0 + lane) * (3 * CC) + 2 * CC + h * HDIM;
      #pragma unroll
      for (int cc = 0; cc < 8; ++cc) {
        v8h vv = *(const v8h*)(vp + cc * 8);
        #pragma unroll
        for (int e = 0; e < 8; ++e) Vt[(cc * 8 + e) * 32 + lane] = vv[e];
      }
    }
    __syncthreads();

    const _Float16* pb = Pt + ln * 32;
    v16h pf = mk16(*(const v8h*)(pb + 16 * h2), *(const v8h*)(pb + 16 * h2 + 8));
    #pragma unroll
    for (int nt = 0; nt < 4; ++nt) {
      const _Float16* vb = Vt + (ln + 16 * nt) * 32 + 16 * h2;
      v16h vf = mk16(*(const v8h*)vb, *(const v8h*)(vb + 8));
      o[nt] = __builtin_amdgcn_wmma_f32_16x16x32_f16(false, pf, false, vf, (short)0, o[nt], false, false);
    }
    __syncthreads();
  }

  #pragma unroll
  for (int r = 0; r < 8; ++r) {
    float rinv = __builtin_amdgcn_rcpf(rsum[r]);
    #pragma unroll
    for (int nt = 0; nt < 4; ++nt) {
      int row = m0 + r + 8 * h2;
      y[(bT + row) * CC + h * HDIM + nt * 16 + ln] = (_Float16)(o[nt][r] * rinv);
    }
  }
}

// ---------------------------------------------------------------------------
extern "C" void kernel_launch(void* const* d_in, const int* in_sizes, int n_in,
                              void* d_out, int out_size, void* d_ws, size_t ws_size,
                              hipStream_t stream) {
  const float* x           = (const float*)d_in[0];
  const float* w_qkv       = (const float*)d_in[1];
  const float* b_qkv       = (const float*)d_in[2];
  const float* w_attn_proj = (const float*)d_in[3];
  const float* b_attn_proj = (const float*)d_in[4];
  const float* w_fc        = (const float*)d_in[5];
  const float* b_fc        = (const float*)d_in[6];
  const float* w_mlp_proj  = (const float*)d_in[7];
  const float* b_mlp_proj  = (const float*)d_in[8];
  const float* ln1_g       = (const float*)d_in[9];
  const float* ln1_b       = (const float*)d_in[10];
  const float* ln2_g       = (const float*)d_in[11];
  const float* ln2_b       = (const float*)d_in[12];

  char* ws = (char*)d_ws;
  size_t off = 0;
  auto alloc = [&](size_t bytes) -> void* {
    void* p = ws + off;
    off = (off + bytes + 255) & ~(size_t)255;
    return p;
  };
  _Float16* wqkv16 = (_Float16*)alloc((size_t)CC * 3 * CC * 2);
  _Float16* wap16  = (_Float16*)alloc((size_t)CC * CC * 2);
  _Float16* wfc16  = (_Float16*)alloc((size_t)CC * 4 * CC * 2);
  _Float16* wmp16  = (_Float16*)alloc((size_t)4 * CC * CC * 2);
  _Float16* xhat1  = (_Float16*)alloc((size_t)MR * CC * 2);
  _Float16* qkv16  = (_Float16*)alloc((size_t)MR * 3 * CC * 2);
  _Float16* y16    = (_Float16*)alloc((size_t)MR * CC * 2);
  float*    resid1 = (float*)   alloc((size_t)MR * CC * 4);
  _Float16* xhat2  = (_Float16*)alloc((size_t)MR * CC * 2);
  _Float16* act16  = (_Float16*)alloc((size_t)MR * 4 * CC * 2);

  cvt_kernel<<<1024, 256, 0, stream>>>(w_qkv, wqkv16, CC * 3 * CC);
  cvt_kernel<<<512, 256, 0, stream>>>(w_attn_proj, wap16, CC * CC);
  cvt_kernel<<<1024, 256, 0, stream>>>(w_fc, wfc16, CC * 4 * CC);
  cvt_kernel<<<1024, 256, 0, stream>>>(w_mlp_proj, wmp16, 4 * CC * CC);

  const float qscale = 0.04419417382415922f;  // 1/sqrt(512)

  ln_kernel<<<MR, 256, 0, stream>>>(x, ln1_g, ln1_b, xhat1);
  gemm_kernel<<<dim3(MR / 128, (3 * CC) / 128), 256, 0, stream>>>(
      xhat1, wqkv16, b_qkv, nullptr, nullptr, qkv16, MR, 3 * CC, CC, 0, qscale, CC);
  attn_kernel<<<dim3(TT / 16, NH, NB), 32, 0, stream>>>(qkv16, y16);
  gemm_kernel<<<dim3(MR / 128, CC / 128), 256, 0, stream>>>(
      y16, wap16, b_attn_proj, x, resid1, nullptr, MR, CC, CC, 0, 1.0f, 0);
  ln_kernel<<<MR, 256, 0, stream>>>(resid1, ln2_g, ln2_b, xhat2);
  gemm_kernel<<<dim3(MR / 128, (4 * CC) / 128), 256, 0, stream>>>(
      xhat2, wfc16, b_fc, nullptr, nullptr, act16, MR, 4 * CC, CC, 1, 1.0f, 0);
  gemm_kernel<<<dim3(MR / 128, CC / 128), 256, 0, stream>>>(
      act16, wmp16, b_mlp_proj, resid1, (float*)d_out, nullptr, MR, CC, 4 * CC, 0, 1.0f, 0);
}
